// SmallDataDecoderViT_82420422410638
// MI455X (gfx1250) — compile-verified, matrix operand order checked by reading, not run on hardware
//
#include <hip/hip_runtime.h>

// ---------------------------------------------------------------------------
// Types for CDNA5 WMMA (wave32, 16x16x32 bf16 -> f32)
// ---------------------------------------------------------------------------
typedef __attribute__((ext_vector_type(16))) __bf16 v16bf;
typedef __attribute__((ext_vector_type(8)))  float  v8f;

union FragU { v16bf v; unsigned u[8]; };

__device__ __forceinline__ unsigned short f2bf(float x) {
    unsigned u = __builtin_bit_cast(unsigned, x);
    unsigned r = (u + 0x7fffu + ((u >> 16) & 1u)) >> 16;
    return (unsigned short)r;
}
__device__ __forceinline__ float bf2f(unsigned short b) {
    unsigned u = ((unsigned)b) << 16;
    return __builtin_bit_cast(float, u);
}

// Load a 16x32 bf16 fragment in the documented A-matrix layout.
// mat is [rows][ld] bf16 (row-major).  lane<16: row=row0+lane, K={0..7,16..23};
// lane>=16: row=row0+lane-16, K={8..15,24..31}.  Row clamped to 0 if >= maxrow
// (caller masks the contribution).
__device__ __forceinline__ v16bf load_frag(const unsigned short* __restrict__ mat,
                                           int ld, int row0, int maxrow, int kb,
                                           int lane) {
    FragU f;
    int r = row0 + (lane & 15);
    if (r >= maxrow) r = 0;
    int half = (lane >> 4) & 1;
    const unsigned short* rp = mat + (long)r * ld + kb + half * 8;
    f.u[0] = *(const unsigned*)(rp + 0);
    f.u[1] = *(const unsigned*)(rp + 2);
    f.u[2] = *(const unsigned*)(rp + 4);
    f.u[3] = *(const unsigned*)(rp + 6);
    f.u[4] = *(const unsigned*)(rp + 16);
    f.u[5] = *(const unsigned*)(rp + 18);
    f.u[6] = *(const unsigned*)(rp + 20);
    f.u[7] = *(const unsigned*)(rp + 22);
    return f.v;
}

#define WMMA_BF16(A, B, C) \
    __builtin_amdgcn_wmma_f32_16x16x32_bf16(false, (A), false, (B), (short)0, (C), false, false)

// ---------------------------------------------------------------------------
// Model constants
// ---------------------------------------------------------------------------
#define BB   8
#define IMG  448
#define PP   16
#define GH   28
#define NN   784
#define PD   256
#define DD   768
#define HH   12
#define HDp  64
#define LL   6
#define FFd  3072
#define NSEC 11
#define ROWS (BB * NN)   /* 6272 */

// ---------------------------------------------------------------------------
// Weight transpose + fp32->bf16 via LDS tile (coalesced both sides):
//   dst[c*R + r] = bf16(src[r*C + c]);  R, C multiples of 32.
// Block = (32, 8); grid = (C/32, R/32).
// ---------------------------------------------------------------------------
__global__ void k_tcvt(const float* __restrict__ src, unsigned short* __restrict__ dst,
                       int R, int C) {
    __shared__ float tile[32][33];
    int c0 = blockIdx.x * 32, r0 = blockIdx.y * 32;
    int tx = threadIdx.x, ty = threadIdx.y;
#pragma unroll
    for (int i = ty; i < 32; i += 8)
        tile[i][tx] = src[(long)(r0 + i) * C + (c0 + tx)];
    __syncthreads();
#pragma unroll
    for (int i = ty; i < 32; i += 8)
        dst[(long)(c0 + i) * R + (r0 + tx)] = f2bf(tile[tx][i]);
}

// ---------------------------------------------------------------------------
// Patchify + LayerNorm(PD) -> bf16 rows (ROWS x PD).  One row per block (256 thr).
// ---------------------------------------------------------------------------
__global__ void k_patchln(const float* __restrict__ x, const float* __restrict__ g,
                          const float* __restrict__ bgain, unsigned short* __restrict__ xp) {
    int row = blockIdx.x;                 // b*784 + n
    int b = row / NN, n = row % NN;
    int gy = n / GH, gx = n % GH;
    int pd = threadIdx.x;                 // 0..255 (C=1)
    int py = pd >> 4, px = pd & 15;
    float v = x[((long)b * IMG + (gy * PP + py)) * IMG + (gx * PP + px)];
    __shared__ float r1[256], r2[256];
    r1[pd] = v; r2[pd] = v * v;
    __syncthreads();
    for (int o = 128; o; o >>= 1) {
        if (pd < o) { r1[pd] += r1[pd + o]; r2[pd] += r2[pd + o]; }
        __syncthreads();
    }
    float mean = r1[0] * (1.0f / PD);
    float var  = r2[0] * (1.0f / PD) - mean * mean;
    float rs = rsqrtf(var + 1e-5f);
    xp[(long)row * PD + pd] = f2bf((v - mean) * rs * g[pd] + bgain[pd]);
}

// ---------------------------------------------------------------------------
// Generic LayerNorm(D) of fp32 rows -> bf16 rows.  One row per block (256 thr).
// ---------------------------------------------------------------------------
__global__ void k_ln(const float* __restrict__ in, int D,
                     const float* __restrict__ g, const float* __restrict__ bgain,
                     unsigned short* __restrict__ out) {
    int row = blockIdx.x;
    const float* rp = in + (long)row * D;
    __shared__ float r1[256], r2[256];
    float s = 0.f, s2 = 0.f;
    for (int i = threadIdx.x; i < D; i += 256) { float v = rp[i]; s += v; s2 += v * v; }
    r1[threadIdx.x] = s; r2[threadIdx.x] = s2;
    __syncthreads();
    for (int o = 128; o; o >>= 1) {
        if (threadIdx.x < o) { r1[threadIdx.x] += r1[threadIdx.x + o]; r2[threadIdx.x] += r2[threadIdx.x + o]; }
        __syncthreads();
    }
    float mean = r1[0] / D;
    float var  = r2[0] / D - mean * mean;
    float rs = rsqrtf(var + 1e-5f);
    for (int i = threadIdx.x; i < D; i += 256)
        out[(long)row * D + i] = f2bf((rp[i] - mean) * rs * g[i] + bgain[i]);
}

// ---------------------------------------------------------------------------
// WMMA GEMM: out(MxN) = A(MxK,bf16) * Bt(NxK,bf16)^T  with fused epilogues.
// Block = 4 waves; each wave computes a 32x64 tile (2 A frags x 4 B frags,
// 8 wmma per 32-k step -> 0.75 fragment loads per wmma).  Block tile 128x64.
//  mode 0: outf = acc + bias[c] + aux[(r%784)*N + c]       (patch embed)
//  mode 1: outh = bf16(acc)                                 (qkv, no bias)
//  mode 2: outf[r,c] += ls[c] * (acc + bias[c])             (residual + layer scale)
//  mode 3: outh = bf16(gelu_exact(acc + bias[c]))           (ff1)
//  mode 4: outf = acc + bias[c]                             (head)
// ---------------------------------------------------------------------------
__global__ void k_gemm(const unsigned short* __restrict__ A, int lda,
                       const unsigned short* __restrict__ Bt, int ldb,
                       int M, int N, int K, int mode,
                       const float* __restrict__ bias,
                       const float* __restrict__ lsp,
                       const float* __restrict__ aux,
                       float* __restrict__ outf, unsigned short* __restrict__ outh) {
    int wave = threadIdx.x >> 5;
    int lane = threadIdx.x & 31;
    int m0 = blockIdx.x * 128 + wave * 32;
    int n0 = blockIdx.y * 64;
    if (m0 >= M) return;

    v8f acc[2][4] = {};
    const unsigned short* Ab0 = A + (long)m0 * lda;
    const unsigned short* Ab1 = A + (long)(m0 + 16) * lda;
    for (int kb = 0; kb < K; kb += 32) {
        v16bf a0 = load_frag(Ab0, lda, 0, 1 << 30, kb, lane);
        v16bf a1 = load_frag(Ab1, lda, 0, 1 << 30, kb, lane);
#pragma unroll
        for (int g = 0; g < 4; ++g) {
            v16bf bfm = load_frag(Bt, ldb, n0 + g * 16, 1 << 30, kb, lane);
            acc[0][g] = WMMA_BF16(a0, bfm, acc[0][g]);
            acc[1][g] = WMMA_BF16(a1, bfm, acc[1][g]);
        }
    }
    int cl = lane & 15;
    int rOff = (lane >> 4) * 8;
#pragma unroll
    for (int mi = 0; mi < 2; ++mi) {
#pragma unroll
        for (int g = 0; g < 4; ++g) {
            int c = n0 + g * 16 + cl;
#pragma unroll
            for (int j = 0; j < 8; ++j) {
                int r = m0 + mi * 16 + rOff + j;
                float vv = acc[mi][g][j];
                long oi = (long)r * N + c;
                if (mode == 0) {
                    outf[oi] = vv + bias[c] + aux[(long)(r % NN) * N + c];
                } else if (mode == 1) {
                    outh[oi] = f2bf(vv);
                } else if (mode == 2) {
                    outf[oi] += lsp[c] * (vv + bias[c]);
                } else if (mode == 3) {
                    float z = vv + bias[c];
                    outh[oi] = f2bf(0.5f * z * (1.0f + erff(z * 0.70710678118f)));
                } else {
                    outf[oi] = vv + bias[c];
                }
            }
        }
    }
}

// ---------------------------------------------------------------------------
// Flash attention (content branch).  One wave per block; block = (b*H+h, qtile).
// qkv rows: [(b*784+n)][2304] bf16, q at col h*64, k at 768+h*64, v at 1536+h*64.
// Writes v_content fp32 (B,H,N,HD).
// ---------------------------------------------------------------------------
__global__ void k_attn(const unsigned short* __restrict__ qkv,
                       float* __restrict__ vout) {
    const int bh = blockIdx.x;            // 0..95
    const int qt = blockIdx.y;            // 0..48
    const int b = bh / HH, h = bh % HH;
    const int lane = threadIdx.x;
    const float scale = 0.125f;           // HD^-0.5

    __shared__ unsigned short Pt[16 * 32];
    __shared__ unsigned short Vt[64 * 32];

    const unsigned short* base = qkv + (long)b * NN * (3 * DD);
    const unsigned short* Qp = base + (long)(qt * 16) * (3 * DD) + h * HDp;
    const unsigned short* Kp = base + DD + h * HDp;
    const unsigned short* Vp = base + 2 * DD + h * HDp;

    // Q fragments are loop-invariant.
    v16bf a0 = load_frag(Qp, 3 * DD, 0, 1 << 30, 0, lane);
    v16bf a1 = load_frag(Qp, 3 * DD, 0, 1 << 30, 32, lane);

    v8f O[4] = {};
    float mrow[8], lrow[8];
#pragma unroll
    for (int j = 0; j < 8; ++j) { mrow[j] = -1e30f; lrow[j] = 0.f; }

    for (int kt = 0; kt < 25; ++kt) {     // 784 = 24*32 + 16 keys
        int k0 = kt * 32;
        // ---- scores: S(16x32) = Q(16x64) @ K^T --------------------------------
        v8f S0 = {}, S1 = {};
        {
            v16bf b00 = load_frag(Kp, 3 * DD, k0,      NN, 0,  lane);
            v16bf b01 = load_frag(Kp, 3 * DD, k0,      NN, 32, lane);
            v16bf b10 = load_frag(Kp, 3 * DD, k0 + 16, NN, 0,  lane);
            v16bf b11 = load_frag(Kp, 3 * DD, k0 + 16, NN, 32, lane);
            S0 = WMMA_BF16(a0, b00, S0);
            S0 = WMMA_BF16(a1, b01, S0);
            S1 = WMMA_BF16(a0, b10, S1);
            S1 = WMMA_BF16(a1, b11, S1);
        }
        // ---- online softmax ---------------------------------------------------
        int colk0 = k0 + (lane & 15);
        bool ok0 = colk0 < NN;
        bool ok1 = (colk0 + 16) < NN;
        int prow_base = (lane >> 4) * 8;
#pragma unroll
        for (int j = 0; j < 8; ++j) {
            float s0 = ok0 ? S0[j] * scale : -1e30f;
            float s1 = ok1 ? S1[j] * scale : -1e30f;
            float mx = fmaxf(s0, s1);
            for (int m = 8; m; m >>= 1) mx = fmaxf(mx, __shfl_xor(mx, m, 32));
            float mnew = fmaxf(mrow[j], mx);
            float alpha = __expf(mrow[j] - mnew);
            float p0 = ok0 ? __expf(s0 - mnew) : 0.f;
            float p1 = ok1 ? __expf(s1 - mnew) : 0.f;
            float ps = p0 + p1;
            for (int m = 8; m; m >>= 1) ps += __shfl_xor(ps, m, 32);
            lrow[j] = lrow[j] * alpha + ps;
            mrow[j] = mnew;
#pragma unroll
            for (int g = 0; g < 4; ++g) O[g][j] *= alpha;
            int pr = (prow_base + j) * 32 + (lane & 15);
            Pt[pr]      = f2bf(p0);
            Pt[pr + 16] = f2bf(p1);
        }
        // ---- stage V tile transposed: Vt[hd][key] -----------------------------
        for (int e = lane; e < 32 * 64; e += 32) {
            int key = e >> 6, hd = e & 63;
            int gk = k0 + key;
            unsigned short vv = (gk < NN) ? Vp[(long)gk * (3 * DD) + hd] : (unsigned short)0;
            Vt[hd * 32 + key] = vv;
        }
        __syncthreads();
        // ---- O += P(16x32) @ V(32x64) -----------------------------------------
        v16bf pa = load_frag(Pt, 32, 0, 16, 0, lane);
#pragma unroll
        for (int g = 0; g < 4; ++g) {
            v16bf vb = load_frag(Vt, 32, g * 16, 64, 0, lane);
            O[g] = WMMA_BF16(pa, vb, O[g]);
        }
        __syncthreads();
    }
    // ---- normalize and store ----------------------------------------------
#pragma unroll
    for (int g = 0; g < 4; ++g) {
        int c = g * 16 + (lane & 15);
#pragma unroll
        for (int j = 0; j < 8; ++j) {
            int rl = ((lane >> 4) << 3) + j;
            vout[((long)bh * NN + qt * 16 + rl) * HDp + c] = O[g][j] / lrow[j];
        }
    }
}

// ---------------------------------------------------------------------------
// Sector means of V: sm[b,h,sec,hd] = mean over {n : sector[n]==sec} of v[b,h,n,hd]
// Grid = B*H*NSEC blocks, 64 threads (hd).
// ---------------------------------------------------------------------------
__global__ void k_secmean(const unsigned short* __restrict__ qkv,
                          const int* __restrict__ sec, float* __restrict__ sm) {
    int idx = blockIdx.x;
    int s = idx % NSEC;
    int bh = idx / NSEC;
    int b = bh / HH, h = bh % HH;
    int hd = threadIdx.x;
    const unsigned short* Vp = qkv + (long)b * NN * (3 * DD) + 2 * DD + h * HDp + hd;
    float acc = 0.f;
    int cnt = 0;
    for (int n = 0; n < NN; ++n) {
        if (sec[n] == s) { acc += bf2f(Vp[(long)n * (3 * DD)]); ++cnt; }
    }
    sm[((long)bh * NSEC + s) * HDp + hd] = acc / fmaxf((float)cnt, 1.0f);
}

// ---------------------------------------------------------------------------
// Gate combine: o = g*v_pos + (1-g)*v_content, reshape (B,H,N,HD)->(B,N,D), bf16.
// ---------------------------------------------------------------------------
__global__ void k_combine(const float* __restrict__ vc, const float* __restrict__ sm,
                          const int* __restrict__ sec, const float* __restrict__ gate,
                          unsigned short* __restrict__ out) {
    long i = (long)blockIdx.x * 256 + threadIdx.x;
    if (i >= (long)ROWS * DD) return;
    int d = (int)(i % DD);
    long bn = i / DD;
    int n = (int)(bn % NN);
    int b = (int)(bn / NN);
    int h = d / HDp, hd = d % HDp;
    float g = 1.0f / (1.0f + __expf(-gate[h]));
    int bh = b * HH + h;
    float vpos = sm[((long)bh * NSEC + sec[n]) * HDp + hd];
    float vcon = vc[((long)bh * NN + n) * HDp + hd];
    out[i] = f2bf(g * vpos + (1.0f - g) * vcon);
}

// ---------------------------------------------------------------------------
// Unpatchify tok (B,N,PD) fp32 -> image (B,1,448,448)
// ---------------------------------------------------------------------------
__global__ void k_unpatch(const float* __restrict__ tok, float* __restrict__ img) {
    long i = (long)blockIdx.x * 256 + threadIdx.x;
    if (i >= (long)BB * IMG * IMG) return;
    int xcol = (int)(i % IMG);
    long t = i / IMG;
    int y = (int)(t % IMG);
    int b = (int)(t / IMG);
    int gy = y >> 4, py = y & 15, gx = xcol >> 4, px = xcol & 15;
    img[i] = tok[((long)b * NN + gy * GH + gx) * PD + py * PP + px];
}

// ---------------------------------------------------------------------------
// Host orchestration
// ---------------------------------------------------------------------------
extern "C" void kernel_launch(void* const* d_in, const int* in_sizes, int n_in,
                              void* d_out, int out_size, void* d_ws, size_t ws_size,
                              hipStream_t stream) {
    (void)in_sizes; (void)n_in; (void)out_size; (void)ws_size;
    const float* x      = (const float*)d_in[0];
    const int*   sector = (const int*)d_in[1];
    const float* peg    = (const float*)d_in[2];
    const float* peb    = (const float*)d_in[3];
    const float* peW    = (const float*)d_in[4];
    const float* pebias = (const float*)d_in[5];
    const float* pos    = (const float*)d_in[6];
    const float* n1g    = (const float*)d_in[7];
    const float* n1b    = (const float*)d_in[8];
    const float* qkvW   = (const float*)d_in[9];
    const float* projW  = (const float*)d_in[10];
    const float* projB  = (const float*)d_in[11];
    const float* gate   = (const float*)d_in[12];
    const float* ls1    = (const float*)d_in[13];
    const float* n2g    = (const float*)d_in[14];
    const float* n2b    = (const float*)d_in[15];
    const float* w1     = (const float*)d_in[16];
    const float* b1     = (const float*)d_in[17];
    const float* w2     = (const float*)d_in[18];
    const float* b2     = (const float*)d_in[19];
    const float* ls2    = (const float*)d_in[20];
    const float* fng    = (const float*)d_in[21];
    const float* fnb    = (const float*)d_in[22];
    const float* headW  = (const float*)d_in[23];
    const float* headB  = (const float*)d_in[24];
    float* out = (float*)d_out;

    char* ws = (char*)d_ws;
    size_t off = 0;
    auto take = [&](size_t bytes) -> char* {
        char* p = ws + off;
        off = (off + bytes + 255) & ~(size_t)255;
        return p;
    };
    // bf16 weights, transposed to [N][K]
    unsigned short* peWt   = (unsigned short*)take((size_t)DD * PD * 2);
    unsigned short* qkvWt  = (unsigned short*)take((size_t)LL * 3 * DD * DD * 2);
    unsigned short* projWt = (unsigned short*)take((size_t)LL * DD * DD * 2);
    unsigned short* ff1Wt  = (unsigned short*)take((size_t)LL * FFd * DD * 2);
    unsigned short* ff2Wt  = (unsigned short*)take((size_t)LL * DD * FFd * 2);
    unsigned short* headWt = (unsigned short*)take((size_t)PD * DD * 2);
    // activations
    unsigned short* xp     = (unsigned short*)take((size_t)ROWS * PD * 2);
    float*          tbuf   = (float*)take((size_t)ROWS * DD * 4);
    unsigned short* hbuf   = (unsigned short*)take((size_t)ROWS * DD * 2);
    unsigned short* qkvbuf = (unsigned short*)take((size_t)ROWS * 3 * DD * 2);
    float*          vcont  = (float*)take((size_t)BB * HH * NN * HDp * 4);
    float*          smbuf  = (float*)take((size_t)BB * HH * NSEC * HDp * 4);
    unsigned short* attin  = (unsigned short*)take((size_t)ROWS * DD * 2);
    unsigned short* ffact  = (unsigned short*)take((size_t)ROWS * FFd * 2);
    float*          tok    = (float*)take((size_t)ROWS * PD * 4);

    auto tcvt = [&](const float* s, unsigned short* d, int R, int C) {
        k_tcvt<<<dim3(C / 32, R / 32), dim3(32, 8), 0, stream>>>(s, d, R, C);
    };
    auto gemm = [&](const unsigned short* A, int lda, const unsigned short* Bt, int ldb,
                    int M, int N, int K, int mode, const float* bias, const float* lsp,
                    const float* aux, float* of, unsigned short* oh) {
        k_gemm<<<dim3(M / 128, N / 64), dim3(128), 0, stream>>>(
            A, lda, Bt, ldb, M, N, K, mode, bias, lsp, aux, of, oh);
    };

    // ---- convert weights ---------------------------------------------------
    tcvt(peW, peWt, PD, DD);                                  // -> [768][256]
    tcvt(headW, headWt, DD, PD);                              // -> [256][768]
    for (int l = 0; l < LL; ++l) {
        tcvt(qkvW  + (long)l * DD * 3 * DD, qkvWt  + (long)l * 3 * DD * DD, DD, 3 * DD);
        tcvt(projW + (long)l * DD * DD,     projWt + (long)l * DD * DD,     DD, DD);
        tcvt(w1    + (long)l * DD * FFd,    ff1Wt  + (long)l * FFd * DD,    DD, FFd);
        tcvt(w2    + (long)l * FFd * DD,    ff2Wt  + (long)l * DD * FFd,    FFd, DD);
    }

    // ---- patch embed -------------------------------------------------------
    k_patchln<<<dim3(ROWS), dim3(256), 0, stream>>>(x, peg, peb, xp);
    gemm(xp, PD, peWt, PD, ROWS, DD, PD, 0, pebias, nullptr, pos, tbuf, nullptr);

    // ---- transformer layers ------------------------------------------------
    for (int l = 0; l < LL; ++l) {
        const unsigned short* qWt = qkvWt  + (long)l * 3 * DD * DD;
        const unsigned short* pWt = projWt + (long)l * DD * DD;
        const unsigned short* f1  = ff1Wt  + (long)l * FFd * DD;
        const unsigned short* f2  = ff2Wt  + (long)l * DD * FFd;

        k_ln<<<dim3(ROWS), dim3(256), 0, stream>>>(tbuf, DD, n1g + l * DD, n1b + l * DD, hbuf);
        gemm(hbuf, DD, qWt, DD, ROWS, 3 * DD, DD, 1, nullptr, nullptr, nullptr, nullptr, qkvbuf);

        k_attn<<<dim3(BB * HH, NN / 16), dim3(32), 0, stream>>>(qkvbuf, vcont);
        k_secmean<<<dim3(BB * HH * NSEC), dim3(HDp), 0, stream>>>(qkvbuf, sector, smbuf);
        {
            long tot = (long)ROWS * DD;
            k_combine<<<dim3((unsigned)((tot + 255) / 256)), dim3(256), 0, stream>>>(
                vcont, smbuf, sector, gate + l * HH, attin);
        }
        gemm(attin, DD, pWt, DD, ROWS, DD, DD, 2, projB + l * DD, ls1 + l * DD,
             nullptr, tbuf, nullptr);

        k_ln<<<dim3(ROWS), dim3(256), 0, stream>>>(tbuf, DD, n2g + l * DD, n2b + l * DD, hbuf);
        gemm(hbuf, DD, f1, DD, ROWS, FFd, DD, 3, b1 + l * FFd, nullptr, nullptr, nullptr, ffact);
        gemm(ffact, FFd, f2, FFd, ROWS, DD, FFd, 2, b2 + l * DD, ls2 + l * DD,
             nullptr, tbuf, nullptr);
    }

    // ---- head --------------------------------------------------------------
    k_ln<<<dim3(ROWS), dim3(256), 0, stream>>>(tbuf, DD, fng, fnb, hbuf);
    gemm(hbuf, DD, headWt, DD, ROWS, PD, DD, 4, headB, nullptr, nullptr, tok, nullptr);
    {
        long tot = (long)BB * IMG * IMG;
        k_unpatch<<<dim3((unsigned)((tot + 255) / 256)), dim3(256), 0, stream>>>(tok, out);
    }
}
